// GBRM_2594160247405
// MI455X (gfx1250) — matrix operand order repaired and checked by reading, not compiled, not run on hardware
//
#include <hip/hip_runtime.h>
#include <math.h>

typedef __bf16 bf16_t;
typedef __attribute__((ext_vector_type(8)))  __bf16 v8bf;
typedef __attribute__((ext_vector_type(16))) __bf16 v16bf;
typedef __attribute__((ext_vector_type(8)))  float  v8f;
typedef __attribute__((ext_vector_type(4)))  float  v4f;

static constexpr int Bn = 16;
static constexpr int Cc = 32;
static constexpr int Hh = 256;
static constexpr int Ww = 256;
static constexpr int HW = Hh * Ww;          // 65536 = 2^16
static constexpr int PLANE = Cc * HW;       // 2^21
static constexpr int TILE_W = 128;
static constexpr int PCOLS = TILE_W + 2;    // 130
static constexpr int TOT = Bn * Cc * HW;    // 33,554,432

// ---------------------------------------------------------------------------
// Weight transform: w[co][ci][kh][kw] f32 -> wt[(kh*3+kw)*32 + co][ci] bf16
// ---------------------------------------------------------------------------
__global__ void wt_transform(const float* __restrict__ w, bf16_t* __restrict__ out, int Cin) {
    int i = blockIdx.x * 256 + threadIdx.x;
    int total = 9 * 32 * Cin;
    if (i >= total) return;
    int ci = i % Cin;
    int t  = i / Cin;
    int co = t % 32;
    int kk = t / 32;
    int kh = kk / 3, kw = kk % 3;
    out[i] = (bf16_t)w[((co * Cin + ci) * 3 + kh) * 3 + kw];
}

// ---------------------------------------------------------------------------
// Fragment loaders (LDS -> VGPR), per ISA 7.12.2 bf16 layouts.
// A 16x32: lane m (0..15), kHalf = lane>>4; elems 0..7 = K ci0+kHalf*8+i,
//          elems 8..15 = K ci0+16+kHalf*8+(i-8)  -> two 16B loads at +0,+16.
// B 32x16: lane n, kHalf; same K order as A -> two 16B loads at +0,+16.
// ---------------------------------------------------------------------------
__device__ __forceinline__ v16bf load_frag_a(const bf16_t* p) {
    v8bf lo = *(const v8bf*)(p);
    v8bf hi = *(const v8bf*)(p + 16);
    return __builtin_shufflevector(lo, hi, 0,1,2,3,4,5,6,7,8,9,10,11,12,13,14,15);
}
__device__ __forceinline__ v16bf load_frag_b(const bf16_t* p) {
    v8bf lo = *(const v8bf*)(p);
    v8bf hi = *(const v8bf*)(p + 16);
    return __builtin_shufflevector(lo, hi, 0,1,2,3,4,5,6,7,8,9,10,11,12,13,14,15);
}

// ---------------------------------------------------------------------------
// Implicit-GEMM 3x3 SAME conv via V_WMMA_F32_16X16X32_BF16.
// 128 threads = 4 waves; each wave: 32 output pixels x 32 output channels
// (4 WMMAs per 32-wide K chunk). CIN is a template constant so the whole
// K loop fully unrolls and all div/mod fold to shifts.
// x0 supplies channels [0,32) with strides (x0_cstride==0 => channel bcast),
// x1 supplies channels [32,64) when CIN==64.
// Dynamic LDS: [ weights 9*32*CINP ][ patch 3*PCOLS*CINP ]  (bf16, CINP=CIN+8
// pad keeps 16B fragment alignment and spreads LDS banks).
// ---------------------------------------------------------------------------
template<int CIN>
__global__ __launch_bounds__(128)
void conv3x3_wmma(const float* __restrict__ x0, int x0_cstride, int x0_bstride,
                  const float* __restrict__ x1,
                  const bf16_t* __restrict__ wt,
                  float* __restrict__ y)
{
    constexpr int CINP = CIN + 8;
    extern __shared__ __align__(16) char smem_raw[];
    bf16_t* wlds  = (bf16_t*)smem_raw;               // 9*32*CINP bf16
    bf16_t* patch = wlds + 9 * 32 * CINP;            // 3*PCOLS*CINP bf16

    const int tid = threadIdx.x;
    const int w0  = blockIdx.x * TILE_W;
    const int h   = blockIdx.y;
    const int b   = blockIdx.z;

    // stage weights into padded LDS layout (coalesced global reads)
    constexpr int wtot = 9 * 32 * CIN;
    for (int i = tid; i < wtot; i += 128) {
        int ci = i & (CIN - 1);
        int rc = i / CIN;                            // (kk*32+co), shift
        wlds[rc * CINP + ci] = wt[i];
    }

    // stage input patch, col-fastest (coalesced), zero halo, f32->bf16
    constexpr int ptot = 3 * CIN * PCOLS;
    for (int i = tid; i < ptot; i += 128) {
        int col = i % PCOLS;
        int t   = i / PCOLS;
        int ci  = t & (CIN - 1);
        int kh  = t / CIN;
        int hh = h + kh - 1;
        int ww = w0 + col - 1;
        float v = 0.f;
        if ((unsigned)hh < (unsigned)Hh && (unsigned)ww < (unsigned)Ww) {
            if (CIN == 32 || ci < 32) {
                v = x0[(size_t)b * x0_bstride + (size_t)ci * x0_cstride + hh * Ww + ww];
            } else {
                v = x1[(size_t)b * PLANE + (size_t)(ci - 32) * HW + hh * Ww + ww];
            }
        }
        patch[((kh * PCOLS) + col) * CINP + ci] = (bf16_t)v;
    }
    __syncthreads();

    const int lane  = tid & 31;
    const int wave  = tid >> 5;          // 0..3
    const int m     = lane & 15;         // A row / B col / D col
    const int kHalf = lane >> 4;         // 0/1 half-K select
    const int wpix  = wave * 32;         // wave's 32-pixel base in tile

    v8f acc00 = {}; v8f acc01 = {}; v8f acc10 = {}; v8f acc11 = {};

    #pragma unroll
    for (int kk = 0; kk < 9; ++kk) {
        const int kh = kk / 3, kw = kk % 3;
        #pragma unroll
        for (int cc = 0; cc < CIN / 32; ++cc) {
            const int ci0 = cc * 32;
            const bf16_t* ar0 = patch + ((kh * PCOLS) + (wpix + m + kw)) * CINP
                                      + ci0 + kHalf * 8;
            const bf16_t* ar1 = ar0 + 16 * CINP;     // pixels wpix+16..wpix+31
            v16bf A0 = load_frag_a(ar0);
            v16bf A1 = load_frag_a(ar1);

            const bf16_t* bp0 = wlds + (size_t)(kk * 32 + m) * CINP
                                     + ci0 + kHalf * 8;
            const bf16_t* bp1 = bp0 + 16 * CINP;     // co 16..31
            v16bf B0 = load_frag_b(bp0);
            v16bf B1 = load_frag_b(bp1);

            acc00 = __builtin_amdgcn_wmma_f32_16x16x32_bf16(false, A0, false, B0,
                                                            (short)0, acc00, false, false);
            acc01 = __builtin_amdgcn_wmma_f32_16x16x32_bf16(false, A0, false, B1,
                                                            (short)0, acc01, false, false);
            acc10 = __builtin_amdgcn_wmma_f32_16x16x32_bf16(false, A1, false, B0,
                                                            (short)0, acc10, false, false);
            acc11 = __builtin_amdgcn_wmma_f32_16x16x32_bf16(false, A1, false, B1,
                                                            (short)0, acc11, false, false);
        }
    }

    // D layout: lane column n = m (co), rows = 8*kHalf + r -> 8 contiguous f32
    const int p0 = w0 + wpix + 8 * kHalf;            // acc0x pixel base
    const int p1 = p0 + 16;                          // acc1x pixel base
    float* yc0 = y + (((size_t)b * 32 + m)      * Hh + h) * Ww;
    float* yc1 = y + (((size_t)b * 32 + 16 + m) * Hh + h) * Ww;
    {
        v4f t0 = {acc00[0], acc00[1], acc00[2], acc00[3]};
        v4f t1 = {acc00[4], acc00[5], acc00[6], acc00[7]};
        *(v4f*)(yc0 + p0)     = t0;  *(v4f*)(yc0 + p0 + 4) = t1;
        v4f t2 = {acc01[0], acc01[1], acc01[2], acc01[3]};
        v4f t3 = {acc01[4], acc01[5], acc01[6], acc01[7]};
        *(v4f*)(yc1 + p0)     = t2;  *(v4f*)(yc1 + p0 + 4) = t3;
        v4f t4 = {acc10[0], acc10[1], acc10[2], acc10[3]};
        v4f t5 = {acc10[4], acc10[5], acc10[6], acc10[7]};
        *(v4f*)(yc0 + p1)     = t4;  *(v4f*)(yc0 + p1 + 4) = t5;
        v4f t6 = {acc11[0], acc11[1], acc11[2], acc11[3]};
        v4f t7 = {acc11[4], acc11[5], acc11[6], acc11[7]};
        *(v4f*)(yc1 + p1)     = t6;  *(v4f*)(yc1 + p1 + 4) = t7;
    }
}

// ---------------------------------------------------------------------------
// BatchNorm over batch statistics
// stats layout: [0..31]=sum [32..63]=sumsq [64..95]=scale [96..127]=shift
// ---------------------------------------------------------------------------
__global__ void zero_stats(float* stats) {
    if (threadIdx.x < 64) stats[threadIdx.x] = 0.f;
}

__global__ void bn_stats(const float* __restrict__ y, float* __restrict__ stats) {
    __shared__ float ssum[256], ssq[256];
    const int c   = blockIdx.x;          // channel
    const int blk = blockIdx.y;          // 0..63
    const int per = (Bn * HW) / 64;      // 16384 elements per block
    float s = 0.f, q = 0.f;
    for (int i = threadIdx.x; i < per; i += 256) {
        int e   = blk * per + i;
        int b   = e >> 16;               // / HW
        int off = e & (HW - 1);
        float v = y[((size_t)b * 32 + c) * HW + off];
        s += v; q += v * v;
    }
    ssum[threadIdx.x] = s; ssq[threadIdx.x] = q;
    __syncthreads();
    for (int st = 128; st > 0; st >>= 1) {
        if (threadIdx.x < st) {
            ssum[threadIdx.x] += ssum[threadIdx.x + st];
            ssq[threadIdx.x]  += ssq[threadIdx.x + st];
        }
        __syncthreads();
    }
    if (threadIdx.x == 0) {
        atomicAdd(&stats[c],      ssum[0]);
        atomicAdd(&stats[32 + c], ssq[0]);
    }
}

__global__ void bn_finalize(const float* __restrict__ g, const float* __restrict__ bb,
                            float* __restrict__ stats) {
    int c = threadIdx.x;
    if (c >= 32) return;
    const float N = (float)(Bn * HW);
    float mu  = stats[c] / N;
    float var = stats[32 + c] / N - mu * mu;
    float sc  = g[c] * rsqrtf(var + 1e-5f);
    stats[64 + c] = sc;
    stats[96 + c] = bb[c] - mu * sc;
}

__global__ void bn_apply(const float* __restrict__ y, const float* __restrict__ stats,
                         float* __restrict__ out, int relu) {
    size_t i = (size_t)blockIdx.x * 256 + threadIdx.x;
    int c = (int)((i >> 16) & 31);
    float v = y[i] * stats[64 + c] + stats[96 + c];
    if (relu) v = fmaxf(v, 0.f);
    out[i] = v;
}

// ---------------------------------------------------------------------------
// Distance-transform pipeline
// ---------------------------------------------------------------------------
__global__ void dt_head(const float* __restrict__ x, const float* __restrict__ wdt,
                        const float* __restrict__ bdt, float* __restrict__ fg) {
    size_t i = (size_t)blockIdx.x * 256 + threadIdx.x;   // over B*HW
    int b   = (int)(i >> 16);
    int off = (int)(i & (HW - 1));
    float acc = bdt[0];
    #pragma unroll
    for (int c = 0; c < 32; ++c)
        acc += x[((size_t)b * 32 + c) * HW + off] * wdt[c];
    float s = 1.f / (1.f + __expf(-acc));
    fg[i] = (floorf(s * 255.f) > 128.f) ? 1.f : 0.f;
}

__global__ void dt_vert(const float* __restrict__ fg, float* __restrict__ g2) {
    int idx = blockIdx.x * 256 + threadIdx.x;            // B*W = 4096
    if (idx >= Bn * Ww) return;
    int b = idx >> 8, w = idx & (Ww - 1);
    const size_t base = (size_t)b * HW + w;
    float c = 1e6f;
    for (int h = 0; h < Hh; ++h) {
        float r = fg[base + (size_t)h * Ww];
        c = (r > 0.5f) ? c + 1.f : 0.f;
        g2[base + (size_t)h * Ww] = c;                   // dist to zero above
    }
    c = 1e6f;
    for (int h = Hh - 1; h >= 0; --h) {
        float r = fg[base + (size_t)h * Ww];
        c = (r > 0.5f) ? c + 1.f : 0.f;
        float mn = fminf(g2[base + (size_t)h * Ww], c);
        g2[base + (size_t)h * Ww] = mn * mn;             // vertical dist^2
    }
}

__global__ void dt_horiz(const float* __restrict__ g2, float* __restrict__ d) {
    __shared__ float row[Ww];
    const size_t base = (size_t)blockIdx.x * Ww;         // blockIdx = b*H + h
    row[threadIdx.x] = g2[base + threadIdx.x];
    __syncthreads();
    float t = (float)threadIdx.x;
    float best = 3.0e38f;
    for (int j = 0; j < Ww; ++j) {
        float off = t - (float)j;
        best = fminf(best, row[j] + off * off);
    }
    d[base + threadIdx.x] = sqrtf(best);
}

__global__ void dt_minmax(const float* __restrict__ d, float* __restrict__ mm) {
    __shared__ float smin[256], smax[256];
    int b = blockIdx.x;
    float lo = 3.0e38f, hi = -3.0e38f;
    for (int i = threadIdx.x; i < HW; i += 256) {
        float v = d[(size_t)b * HW + i];
        lo = fminf(lo, v); hi = fmaxf(hi, v);
    }
    smin[threadIdx.x] = lo; smax[threadIdx.x] = hi;
    __syncthreads();
    for (int st = 128; st > 0; st >>= 1) {
        if (threadIdx.x < st) {
            smin[threadIdx.x] = fminf(smin[threadIdx.x], smin[threadIdx.x + st]);
            smax[threadIdx.x] = fmaxf(smax[threadIdx.x], smax[threadIdx.x + st]);
        }
        __syncthreads();
    }
    if (threadIdx.x == 0) { mm[b] = smin[0]; mm[16 + b] = smax[0]; }
}

__global__ void dt_norm(const float* __restrict__ d, const float* __restrict__ mm,
                        float* __restrict__ dtn) {
    size_t i = (size_t)blockIdx.x * 256 + threadIdx.x;   // B*HW
    int b = (int)(i >> 16);
    float lo = mm[b], hi = mm[16 + b];
    float rng = hi - lo;
    dtn[i] = (rng > 1e-8f) ? (d[i] - lo) / rng : 0.f;
}

// ---------------------------------------------------------------------------
// Elementwise
// ---------------------------------------------------------------------------
__global__ void ew_gate(const float* __restrict__ a, const float* __restrict__ r,
                        const float* __restrict__ dtn, float* __restrict__ o) {
    size_t i = (size_t)blockIdx.x * 256 + threadIdx.x;   // B*32*HW
    int b = (int)(i >> 21);                              // / (32*HW)
    size_t off = i & (size_t)(HW - 1);
    o[i] = a[i] * r[i] + dtn[((size_t)b << 16) + off];
}

__global__ void ew_add4(const float* __restrict__ p0, const float* __restrict__ p1,
                        const float* __restrict__ p2, const float* __restrict__ p3,
                        float* __restrict__ o) {
    size_t i = (size_t)blockIdx.x * 256 + threadIdx.x;
    o[i] = p0[i] + p1[i] + p2[i] + p3[i];
}

// ---------------------------------------------------------------------------
// Host-side helpers
// ---------------------------------------------------------------------------
static void run_conv_bn(const float* x0, int x0_cs, int x0_bs, const float* x1,
                        const bf16_t* wt, int Cin,
                        const float* g, const float* bvec,
                        float* Y, float* stats, float* out, int relu, hipStream_t s)
{
    dim3 grid(Ww / TILE_W, Hh, Bn);
    if (Cin == 64) {
        constexpr int smem = (9 * 32 * (64 + 8) + 3 * PCOLS * (64 + 8)) * (int)sizeof(bf16_t);
        conv3x3_wmma<64><<<grid, 128, smem, s>>>(x0, x0_cs, x0_bs, x1, wt, Y);
    } else {
        constexpr int smem = (9 * 32 * (32 + 8) + 3 * PCOLS * (32 + 8)) * (int)sizeof(bf16_t);
        conv3x3_wmma<32><<<grid, 128, smem, s>>>(x0, x0_cs, x0_bs, x1, wt, Y);
    }
    zero_stats<<<1, 64, 0, s>>>(stats);
    bn_stats<<<dim3(32, 64), 256, 0, s>>>(Y, stats);
    bn_finalize<<<1, 32, 0, s>>>(g, bvec, stats);
    bn_apply<<<TOT / 256, 256, 0, s>>>(Y, stats, out, relu);
}

static void run_dt(const float* x, const float* wdt, const float* bdt,
                   float* fg, float* g2, float* dbuf, float* mm, float* dtn,
                   hipStream_t s)
{
    dt_head<<<(Bn * HW) / 256, 256, 0, s>>>(x, wdt, bdt, fg);
    dt_vert<<<(Bn * Ww) / 256, 256, 0, s>>>(fg, g2);
    dt_horiz<<<Bn * Hh, Ww, 0, s>>>(g2, dbuf);
    dt_minmax<<<Bn, 256, 0, s>>>(dbuf, mm);
    dt_norm<<<(Bn * HW) / 256, 256, 0, s>>>(dbuf, mm, dtn);
}

extern "C" void kernel_launch(void* const* d_in, const int* in_sizes, int n_in,
                              void* d_out, int out_size, void* d_ws, size_t ws_size,
                              hipStream_t stream) {
    const float* of1   = (const float*)d_in[0];
    const float* xr    = (const float*)d_in[1];
    const float* of2   = (const float*)d_in[2];
    const float* w_cb  = (const float*)d_in[3];
    const float* g_cb  = (const float*)d_in[4];
    const float* b_cb  = (const float*)d_in[5];
    const float* w_cbs = (const float*)d_in[6];
    const float* g_cbs = (const float*)d_in[7];
    const float* b_cbs = (const float*)d_in[8];
    const float* w_dt  = (const float*)d_in[9];
    const float* b_dt  = (const float*)d_in[10];

    // workspace carve-up (256-B aligned slots)
    char* ws = (char*)d_ws;
    size_t o = 0;
    auto carve = [&](size_t bytes) -> char* {
        char* p = ws + o;
        o = (o + bytes + 255) & ~(size_t)255;
        return p;
    };
    float*  stats = (float*)carve(128 * sizeof(float));
    bf16_t* wtcb  = (bf16_t*)carve((size_t)9 * 32 * 64 * sizeof(bf16_t));
    bf16_t* wtcbs = (bf16_t*)carve((size_t)9 * 32 * 32 * sizeof(bf16_t));
    float*  mm    = (float*)carve(32 * sizeof(float));
    float*  fg    = (float*)carve((size_t)Bn * HW * sizeof(float));
    float*  g2    = (float*)carve((size_t)Bn * HW * sizeof(float));
    float*  dbuf  = (float*)carve((size_t)Bn * HW * sizeof(float));
    float*  dtn   = (float*)carve((size_t)Bn * HW * sizeof(float));
    float*  Y     = (float*)carve((size_t)TOT * sizeof(float));
    float*  S0    = (float*)carve((size_t)TOT * sizeof(float));
    float*  S1    = (float*)carve((size_t)TOT * sizeof(float));
    float*  S2    = (float*)carve((size_t)TOT * sizeof(float));
    float*  S3    = (float*)carve((size_t)TOT * sizeof(float));
    if (o > ws_size) return;   // workspace too small: bail deterministically

    float* out = (float*)d_out;

    // weight pre-transforms (f32 OIHW -> bf16 [kk][co][ci])
    wt_transform<<<(9 * 32 * 64 + 255) / 256, 256, 0, stream>>>(w_cb,  wtcb,  64);
    wt_transform<<<(9 * 32 * 32 + 255) / 256, 256, 0, stream>>>(w_cbs, wtcbs, 32);

    // fr = cb(feat_r, feat_of1)                       -> S0
    run_conv_bn(xr, HW, PLANE, of1, wtcb, 64, g_cb, b_cb, Y, stats, S0, 0, stream);
    // fr = cb(fr, feat_of2)                           -> S1   (S0 dead after)
    run_conv_bn(S0, HW, PLANE, of2, wtcb, 64, g_cb, b_cb, Y, stats, S1, 0, stream);
    // dt_r = distance_weight(head(fr))                -> dtn  (S1 dead after)
    run_dt(S1, w_dt, b_dt, fg, g2, dbuf, mm, dtn, stream);
    // o1_fr = cb(dt_r bcast, feat_of1)                -> S0
    run_conv_bn(dtn, 0, HW, of1, wtcb, 64, g_cb, b_cb, Y, stats, S0, 0, stream);
    // o2_fr = cb(dt_r bcast, feat_of2)                -> S1
    run_conv_bn(dtn, 0, HW, of2, wtcb, 64, g_cb, b_cb, Y, stats, S1, 0, stream);
    // o1_cbr = relu(cbs(feat_of1))                    -> S2
    run_conv_bn(of1, HW, PLANE, nullptr, wtcbs, 32, g_cbs, b_cbs, Y, stats, S2, 1, stream);
    // dt_o1 = distance_weight(head(o1_cbr))           -> dtn
    run_dt(S2, w_dt, b_dt, fg, g2, dbuf, mm, dtn, stream);
    // S3 = o1_cbr * feat_r + dt_o1(bcast)
    ew_gate<<<TOT / 256, 256, 0, stream>>>(S2, xr, dtn, S3);
    // o1_sum = cb(gated+dt, feat_r)                   -> S2
    run_conv_bn(S3, HW, PLANE, xr, wtcb, 64, g_cb, b_cb, Y, stats, S2, 0, stream);
    // o1_sum = cb(o1_sum, feat_of2)                   -> S3
    run_conv_bn(S2, HW, PLANE, of2, wtcb, 64, g_cb, b_cb, Y, stats, S3, 0, stream);
    // o2 branch before its second assignment is dead code (o2_sum overwritten):
    // o2_sum = cb(o1_sum, feat_of1)                   -> S2
    run_conv_bn(S3, HW, PLANE, of1, wtcb, 64, g_cb, b_cb, Y, stats, S2, 0, stream);
    // final_sum = o2_sum + o1_fr + o2_fr + o1_sum     -> S0
    ew_add4<<<TOT / 256, 256, 0, stream>>>(S2, S0, S1, S3, S0);
    // out = relu(cbs(final_sum))                      -> d_out
    run_conv_bn(S0, HW, PLANE, nullptr, wtcbs, 32, g_cbs, b_cbs, Y, stats, out, 1, stream);
}
